// Gemma4TextAttention_30288109371941
// MI455X (gfx1250) — matrix operand order repaired and checked by reading, not compiled, not run on hardware
//
#include <hip/hip_runtime.h>

#define S_LEN 2048
#define HID   2816
#define NH    16
#define NKV   8
#define HD    256
#define WIN   1024
#define BATCH 2

typedef __attribute__((ext_vector_type(16))) __bf16 v16bf;
typedef __attribute__((ext_vector_type(8)))  float  v8f;
typedef __attribute__((ext_vector_type(4)))  unsigned int u32x4;
typedef __attribute__((ext_vector_type(8)))  int i32x8;
typedef __attribute__((ext_vector_type(4)))  int i32x4;

union FragU { v16bf v; u32x4 u[2]; };

__device__ __forceinline__ v16bf load_frag2(const __bf16* p0, const __bf16* p1) {
  FragU f;
  f.u[0] = *(const u32x4*)p0;
  f.u[1] = *(const u32x4*)p1;
  return f.v;
}

// ---- Tensor Data Mover: 2D tile (tile_w elems x tile_h rows, 2-byte elems) -> LDS
__device__ __forceinline__ void tdm_load_2d(const void* gaddr, void* lds,
                                            int tile_w, int tile_h,
                                            int stride, int dim0, int dim1) {
  unsigned long long ga = (unsigned long long)gaddr;
  u32x4 g0;
  g0[0] = 1u;                                              // count=1, user mode
  g0[1] = (unsigned)(size_t)lds;                           // lds_addr (bytes)
  g0[2] = (unsigned)(ga & 0xffffffffu);                    // global_addr[31:0]
  g0[3] = (unsigned)((ga >> 32) & 0x01ffffffu) | (2u << 30); // addr[56:32] | type=2
  i32x8 g1;
  g1[0] = (1 << 16);                                       // data_size=1 -> 2 bytes
  g1[1] = (dim0 & 0xffff) << 16;                           // tensor_dim0[15:0]
  g1[2] = ((dim0 >> 16) & 0xffff) | ((dim1 & 0xffff) << 16);
  g1[3] = ((dim1 >> 16) & 0xffff) | ((tile_w & 0xffff) << 16); // tile_dim0
  g1[4] = (tile_h & 0xffff);                               // tile_dim1 (tile_dim2=0)
  g1[5] = stride;                                          // tensor_dim0_stride[31:0]
  g1[6] = 0;
  g1[7] = 0;
  i32x4 z4 = {0, 0, 0, 0};
#if defined(__clang_major__) && (__clang_major__ >= 23)
  i32x8 z8 = {0, 0, 0, 0, 0, 0, 0, 0};
  __builtin_amdgcn_tensor_load_to_lds(g0, g1, z4, z4, z8, 0);
#else
  __builtin_amdgcn_tensor_load_to_lds(g0, g1, z4, z4, 0);
#endif
}

// ---- async global -> LDS, 16 bytes per lane (portable inline asm form)
__device__ __forceinline__ void async_ld128(void* lds, const void* g) {
  asm volatile("global_load_async_to_lds_b128 %0, %1, off"
               :: "v"((unsigned)(size_t)lds), "v"(g) : "memory");
}
__device__ __forceinline__ void wait_async0() {
  asm volatile("s_wait_asynccnt 0x0" ::: "memory");
}

// ---------------------------------------------------------------- fp32 -> bf16
__global__ __launch_bounds__(256)
void cast_bf16_kernel(const float* __restrict__ x, __bf16* __restrict__ y, long n) {
  long i = (long)blockIdx.x * blockDim.x + threadIdx.x;
  long st = (long)gridDim.x * blockDim.x;
  for (; i < n; i += st) y[i] = (__bf16)x[i];
}

// ---------------------------------------------------------------- GEMM (A[M,K] @ Bw[N,K]^T -> C[M,N] fp32)
// 128x128 block tile, 8 waves, each wave 2x4 16x16 WMMA tiles, K step 32.
// TDM double-buffered LDS staging: wave 0 issues two descriptors per step.
__global__ __launch_bounds__(256)
void gemm_nt_bf16(const __bf16* __restrict__ A, const __bf16* __restrict__ Bw,
                  float* __restrict__ C, int M, int N, int K) {
  __shared__ __align__(32) __bf16 As[2][128 * 32];
  __shared__ __align__(32) __bf16 Bs[2][128 * 32];
  const int tid  = threadIdx.x;
  const int wave = tid >> 5, lane = tid & 31;
  const int bm = blockIdx.y * 128, bn = blockIdx.x * 128;
  const int wm = (wave >> 1) * 32, wn = (wave & 1) * 64;
  const int lm   = lane & 15;
  const int kb8  = (lane & 16) ? 8 : 0;
  const int kb16 = (lane & 16) ? 16 : 0;

  v8f acc[2][4] = {};

  if (wave == 0) {
    tdm_load_2d(&A[(size_t)bm * K],  &As[0][0], 32, 128, K, K, M);
    tdm_load_2d(&Bw[(size_t)bn * K], &Bs[0][0], 32, 128, K, K, N);
  }

  int cur = 0;
  for (int k0 = 0; k0 < K; k0 += 32) {
    const bool more = (k0 + 32) < K;
    if (wave == 0) {
      if (more) {
        tdm_load_2d(&A[(size_t)bm * K + k0 + 32],  &As[cur ^ 1][0], 32, 128, K, K, M);
        tdm_load_2d(&Bw[(size_t)bn * K + k0 + 32], &Bs[cur ^ 1][0], 32, 128, K, K, N);
        __builtin_amdgcn_s_wait_tensorcnt(2);   // oldest tile (cur) complete
      } else {
        __builtin_amdgcn_s_wait_tensorcnt(0);
      }
    }
    __syncthreads();
    v16bf af[2], bfx[4];
    for (int mt = 0; mt < 2; ++mt) {
      const __bf16* r = &As[cur][(wm + mt * 16 + lm) * 32];
      af[mt] = load_frag2(r + kb16, r + kb16 + 8);
    }
    for (int nt = 0; nt < 4; ++nt) {
      const __bf16* r = &Bs[cur][(wn + nt * 16 + lm) * 32];
      bfx[nt] = *(const v16bf*)(r + kb16);
    }
    for (int mt = 0; mt < 2; ++mt)
      for (int nt = 0; nt < 4; ++nt)
        acc[mt][nt] = __builtin_amdgcn_wmma_f32_16x16x32_bf16(
            false, af[mt], false, bfx[nt], (short)0, acc[mt][nt], false, false);
    __syncthreads();
    cur ^= 1;
  }
  const int roff = (lane & 16) ? 8 : 0;
  for (int mt = 0; mt < 2; ++mt)
    for (int nt = 0; nt < 4; ++nt)
      for (int r = 0; r < 8; ++r)
        C[(size_t)(bm + wm + mt * 16 + roff + r) * N + (bn + wn + nt * 16 + lm)] =
            acc[mt][nt][r];
}

// ---------------------------------------------------------------- RMSNorm (+ optional RoPE), fp32 in -> bf16 out
__global__ __launch_bounds__(256)
void rmsnorm_rope_kernel(const float* __restrict__ X, __bf16* __restrict__ Y,
                         const float* __restrict__ w, const float* __restrict__ cosb,
                         const float* __restrict__ sinb, int nheads, int use_rope) {
  const int row = blockIdx.x;          // (b*S + s)*nheads + h
  const int bs  = row / nheads;        // b*S + s
  const int d   = threadIdx.x;         // 0..255
  const float* x = X + (size_t)row * HD;
  const float v = x[d];
  __shared__ float red[HD];
  red[d] = v * v;
  __syncthreads();
  for (int s = 128; s > 0; s >>= 1) {
    if (d < s) red[d] += red[d + s];
    __syncthreads();
  }
  const float inv = rsqrtf(red[0] * (1.0f / HD) + 1e-6f);
  float y = v * inv * (w ? w[d] : 1.0f);
  if (use_rope) {
    const int pd = (d < HD / 2) ? d + HD / 2 : d - HD / 2;
    float xp = x[pd] * inv * (w ? w[pd] : 1.0f);
    float rot = (d < HD / 2) ? -xp : xp;
    const float c  = cosb[(size_t)bs * HD + d];
    const float sn = sinb[(size_t)bs * HD + d];
    y = y * c + rot * sn;
  }
  Y[(size_t)row * HD + d] = (__bf16)y;
}

// ---------------------------------------------------------------- V transpose: [b,s,kh,d] -> [b,kh,d,s]
// Tile staged with async global->LDS (no VGPR round trip), then scattered out.
__global__ __launch_bounds__(256)
void transpose_v_kernel(const __bf16* __restrict__ V, __bf16* __restrict__ Vt) {
  __shared__ __align__(16) __bf16 t[32][32];
  const int s0 = blockIdx.x * 32, d0 = blockIdx.y * 32;
  const int bk = blockIdx.z;
  const int b = bk >> 3, kh = bk & 7;
  const int tid = threadIdx.x;
  if (tid < 128) {                                   // waves 0..3 fully active
    int row = tid >> 2, cc = (tid & 3) * 8;          // 128 chunks of 8 bf16
    async_ld128(&t[row][cc],
                &V[(((size_t)b * S_LEN + s0 + row) * NKV + kh) * HD + d0 + cc]);
  }
  wait_async0();
  __syncthreads();
  const int tx = tid & 31, ty = tid >> 5;            // 32 x 8
  for (int i = 0; i < 4; ++i) {
    int d = d0 + ty + i * 8;
    Vt[(((size_t)b * NKV + kh) * HD + d) * S_LEN + s0 + tx] = t[tx][ty + i * 8];
  }
}

// ---------------------------------------------------------------- flash attention, sliding window
__global__ __launch_bounds__(128)
void attn_kernel(const __bf16* __restrict__ Q, const __bf16* __restrict__ K,
                 const __bf16* __restrict__ Vt, __bf16* __restrict__ O) {
  const int QG = S_LEN / 64;                    // 32 query groups
  const int blk = blockIdx.x;
  const int qg = blk % QG;
  const int h  = (blk / QG) % NH;
  const int b  = blk / (QG * NH);
  const int kh = h >> 1;                        // GQA: NH/NKV = 2
  const int wave = threadIdx.x >> 5;
  const int lane = threadIdx.x & 31;
  const int lm   = lane & 15;
  const int hi16 = (lane & 16) ? 1 : 0;
  const int kb8  = hi16 ? 8 : 0;
  const int kb16 = hi16 ? 16 : 0;
  const int q0 = qg * 64 + wave * 16;

  __shared__ __align__(32) __bf16 prob[4][16 * 32];
  __bf16* Pr = prob[wave];

  const __bf16* Qb = Q  + ((size_t)b * S_LEN * NH  + h)  * HD;
  const __bf16* Kb = K  + ((size_t)b * S_LEN * NKV + kh) * HD;
  const __bf16* Vb = Vt + (((size_t)b * NKV + kh) * HD) * S_LEN;

  v8f acc[16] = {};
  float mrow[8], lrow[8];
  for (int r = 0; r < 8; ++r) { mrow[r] = -1e30f; lrow[r] = 0.0f; }

  const int qlo = qg * 64;
  int kfirst = qlo - (WIN - 1); if (kfirst < 0) kfirst = 0;
  kfirst &= ~31;
  const int klast = (qlo + 63) & ~31;

  for (int kb = kfirst; kb <= klast; kb += 32) {
    if (kb + 32 <= klast) {                     // prefetch next key block
      __builtin_prefetch(Kb + (size_t)(kb + 32 + lm) * (NKV * HD), 0, 0);
      __builtin_prefetch(Vb + (size_t)lm * S_LEN + kb + 32, 0, 0);
    }
    // ---- scores: two 16x16 tiles over keys kb..kb+31, K = HD = 256
    v8f sc[2] = {};
    for (int ks = 0; ks < HD; ks += 32) {
      const __bf16* qr = Qb + (size_t)(q0 + lm) * (NH * HD) + ks;
      v16bf aq = load_frag2(qr + kb16, qr + kb16 + 8);
      for (int t = 0; t < 2; ++t) {
        const __bf16* kr = Kb + (size_t)(kb + t * 16 + lm) * (NKV * HD) + ks + kb16;
        v16bf bk = *(const v16bf*)kr;
        sc[t] = __builtin_amdgcn_wmma_f32_16x16x32_bf16(false, aq, false, bk,
                                                        (short)0, sc[t], false, false);
      }
    }
    // ---- masked online softmax
    float p0[8], p1[8];
    for (int r = 0; r < 8; ++r) {
      const int i = q0 + r + hi16 * 8;
      const int j0 = kb + lm, j1 = kb + 16 + lm;
      float s0 = (j0 <= i && (i - j0) < WIN) ? sc[0][r] : -1e30f;
      float s1 = (j1 <= i && (i - j1) < WIN) ? sc[1][r] : -1e30f;
      float cur = fmaxf(s0, s1);
      cur = fmaxf(cur, __shfl_xor(cur, 1, 32));
      cur = fmaxf(cur, __shfl_xor(cur, 2, 32));
      cur = fmaxf(cur, __shfl_xor(cur, 4, 32));
      cur = fmaxf(cur, __shfl_xor(cur, 8, 32));
      const float mn = fmaxf(mrow[r], cur);
      const float scale = __expf(mrow[r] - mn);
      const float e0 = __expf(s0 - mn), e1 = __expf(s1 - mn);
      float rs = e0 + e1;
      rs += __shfl_xor(rs, 1, 32);
      rs += __shfl_xor(rs, 2, 32);
      rs += __shfl_xor(rs, 4, 32);
      rs += __shfl_xor(rs, 8, 32);
      lrow[r] = lrow[r] * scale + rs;
      mrow[r] = mn;
      for (int nt = 0; nt < 16; ++nt) acc[nt][r] *= scale;
      p0[r] = e0; p1[r] = e1;
    }
    for (int r = 0; r < 8; ++r) {
      Pr[(r + hi16 * 8) * 32 + lm]      = (__bf16)p0[r];
      Pr[(r + hi16 * 8) * 32 + 16 + lm] = (__bf16)p1[r];
    }
    __syncthreads();
    // ---- O += P @ V
    const __bf16* pr = &Pr[lm * 32];
    v16bf ap = load_frag2(pr + kb16, pr + kb16 + 8);
    for (int nt = 0; nt < 16; ++nt) {
      const __bf16* vr = Vb + (size_t)(nt * 16 + lm) * S_LEN + kb + kb16;
      v16bf bv = *(const v16bf*)vr;
      acc[nt] = __builtin_amdgcn_wmma_f32_16x16x32_bf16(false, ap, false, bv,
                                                        (short)0, acc[nt], false, false);
    }
    __syncthreads();
  }
  __bf16* Ob = O + ((size_t)b * S_LEN * NH + h) * HD;
  for (int r = 0; r < 8; ++r) {
    const float inv = 1.0f / lrow[r];
    const int qi = q0 + r + hi16 * 8;
    for (int nt = 0; nt < 16; ++nt)
      Ob[(size_t)qi * (NH * HD) + nt * 16 + lm] = (__bf16)(acc[nt][r] * inv);
  }
}

// ---------------------------------------------------------------- host
extern "C" void kernel_launch(void* const* d_in, const int* in_sizes, int n_in,
                              void* d_out, int out_size, void* d_ws, size_t ws_size,
                              hipStream_t stream) {
  (void)in_sizes; (void)n_in; (void)out_size; (void)ws_size;
  const float* hs   = (const float*)d_in[0];
  const float* Wq   = (const float*)d_in[1];
  const float* Wk   = (const float*)d_in[2];
  const float* Wv   = (const float*)d_in[3];
  const float* Wo   = (const float*)d_in[4];
  const float* qw   = (const float*)d_in[5];
  const float* kw   = (const float*)d_in[6];
  const float* cosb = (const float*)d_in[7];
  const float* sinb = (const float*)d_in[8];

  const long M    = (long)BATCH * S_LEN;  // 4096
  const long NQ   = (long)NH * HD;        // 4096
  const long NKVD = (long)NKV * HD;       // 2048

  char* p = (char*)d_ws;
  auto alloc = [&](size_t bytes) -> char* {
    char* r = p; p += (bytes + 255) & ~(size_t)255; return r;
  };
  __bf16* Xb  = (__bf16*)alloc((size_t)M * HID * 2);
  __bf16* Wqb = (__bf16*)alloc((size_t)NQ * HID * 2);
  __bf16* Wkb = (__bf16*)alloc((size_t)NKVD * HID * 2);
  __bf16* Wvb = (__bf16*)alloc((size_t)NKVD * HID * 2);
  __bf16* Wob = (__bf16*)alloc((size_t)HID * NQ * 2);
  float*  Qf  = (float*) alloc((size_t)M * NQ * 4);
  float*  Kf  = (float*) alloc((size_t)M * NKVD * 4);
  float*  Vf  = (float*) alloc((size_t)M * NKVD * 4);
  __bf16* Qb  = (__bf16*)alloc((size_t)M * NQ * 2);
  __bf16* Kb  = (__bf16*)alloc((size_t)M * NKVD * 2);
  __bf16* Vb  = (__bf16*)alloc((size_t)M * NKVD * 2);
  __bf16* Vtb = (__bf16*)alloc((size_t)M * NKVD * 2);
  __bf16* Ab  = (__bf16*)alloc((size_t)M * NQ * 2);

  cast_bf16_kernel<<<2048, 256, 0, stream>>>(hs, Xb,  M * HID);
  cast_bf16_kernel<<<2048, 256, 0, stream>>>(Wq, Wqb, NQ * HID);
  cast_bf16_kernel<<<2048, 256, 0, stream>>>(Wk, Wkb, NKVD * HID);
  cast_bf16_kernel<<<2048, 256, 0, stream>>>(Wv, Wvb, NKVD * HID);
  cast_bf16_kernel<<<2048, 256, 0, stream>>>(Wo, Wob, (long)HID * NQ);

  gemm_nt_bf16<<<dim3(NQ / 128,   M / 128), 256, 0, stream>>>(Xb, Wqb, Qf, (int)M, (int)NQ,   HID);
  gemm_nt_bf16<<<dim3(NKVD / 128, M / 128), 256, 0, stream>>>(Xb, Wkb, Kf, (int)M, (int)NKVD, HID);
  gemm_nt_bf16<<<dim3(NKVD / 128, M / 128), 256, 0, stream>>>(Xb, Wvb, Vf, (int)M, (int)NKVD, HID);

  rmsnorm_rope_kernel<<<(int)(M * NH),  256, 0, stream>>>(Qf, Qb, qw, cosb, sinb, NH, 1);
  rmsnorm_rope_kernel<<<(int)(M * NKV), 256, 0, stream>>>(Kf, Kb, kw, cosb, sinb, NKV, 1);
  rmsnorm_rope_kernel<<<(int)(M * NKV), 256, 0, stream>>>(Vf, Vb, nullptr, nullptr, nullptr, NKV, 0);

  transpose_v_kernel<<<dim3(S_LEN / 32, HD / 32, BATCH * NKV), 256, 0, stream>>>(Vb, Vtb);

  attn_kernel<<<BATCH * NH * (S_LEN / 64), 128, 0, stream>>>(Qb, Kb, Vtb, Ab);

  gemm_nt_bf16<<<dim3(HID / 128, M / 128), 256, 0, stream>>>(Ab, Wob, (float*)d_out,
                                                             (int)M, HID, (int)NQ);
}
